// TwinTowerGCN_64055142252968
// MI455X (gfx1250) — matrix operand order repaired and checked by reading, not compiled, not run on hardware
//
#include <hip/hip_runtime.h>
#include <hip/hip_fp16.h>

// Problem constants (match reference)
#define NN      50000
#define EE      800000
#define IN_DIM  128
#define OUT1    64
#define OUT2    128
#define FILLW   2.0f
#define WPAD    8           // f16 pad on transposed-W row stride: 16B -> distinct LDS banks

typedef __attribute__((ext_vector_type(16))) _Float16 v16h;
typedef __attribute__((ext_vector_type(8)))  _Float16 v8h;
typedef __attribute__((ext_vector_type(8)))  float    v8f;

// ---------------------------------------------------------------------------
// Float atomic max via sign-aware integer atomics (order independent).
// Init value -inf = 0xFF800000 works for both paths.
// ---------------------------------------------------------------------------
__device__ __forceinline__ void atomic_max_f32(float* addr, float v) {
    if (v >= 0.0f) {
        atomicMax((int*)addr, __float_as_int(v));
    } else {
        atomicMin((unsigned int*)addr, __float_as_uint(v));
    }
}

// ---------------------------------------------------------------------------
// Small utility kernels
// ---------------------------------------------------------------------------
__global__ void k_fill_f32(float* __restrict__ p, float v, int n) {
    int i = blockIdx.x * blockDim.x + threadIdx.x;
    if (i < n) p[i] = v;
}

__global__ void k_accum_deg(const int* __restrict__ dst, const float* __restrict__ ew,
                            float* __restrict__ deg, int e) {
    int i = blockIdx.x * blockDim.x + threadIdx.x;
    if (i < e) {
        float w = ew[i];
        w = w > 0.0f ? w : 0.0f;          // relu(edge_weight)
        atomicAdd(&deg[dst[i]], w);
    }
}

__global__ void k_dinv_self(const float* __restrict__ deg, float* __restrict__ dinv,
                            float* __restrict__ selfnorm, int n) {
    int i = blockIdx.x * blockDim.x + threadIdx.x;
    if (i < n) {
        float d = deg[i];
        float r = d > 0.0f ? rsqrtf(d) : 0.0f;
        dinv[i] = r;
        selfnorm[i] = FILLW * r * r;       // self-loop message scale
    }
}

__global__ void k_edge_norm(const int* __restrict__ src, const int* __restrict__ dst,
                            const float* __restrict__ ew, const float* __restrict__ dinv,
                            float* __restrict__ nrm, int e) {
    int i = blockIdx.x * blockDim.x + threadIdx.x;
    if (i < e) {
        float w = ew[i];
        w = w > 0.0f ? w : 0.0f;
        nrm[i] = dinv[src[i]] * w * dinv[dst[i]];
    }
}

__global__ void k_f32_to_f16(const float* __restrict__ s, _Float16* __restrict__ d, int n) {
    int i = blockIdx.x * blockDim.x + threadIdx.x;
    if (i < n) d[i] = (_Float16)s[i];
}

// Transpose f32 W[K x NOUT] -> padded f16 Wt[NOUT x (K+WPAD)] (B operand is
// read K-contiguous per output column; pad kills LDS bank conflicts).
__global__ void k_w_transpose_f16(const float* __restrict__ W, _Float16* __restrict__ Wt,
                                  int K, int NOUT) {
    int i = blockIdx.x * blockDim.x + threadIdx.x;
    int stride = K + WPAD;
    if (i >= NOUT * stride) return;
    int n = i / stride;
    int k = i - n * stride;
    Wt[i] = (k < K) ? (_Float16)W[(size_t)k * NOUT + n] : (_Float16)0.0f;
}

// ---------------------------------------------------------------------------
// WMMA GEMM: C[M x NOUT] = A[M x K] (f16 row major) * W (f16, pre-transposed
// to [NOUT][K+WPAD]), f32 accumulate. One wave32 per 16x16 C tile; K and NOUT
// are template constants so the K loop fully unrolls (multiple back-to-back
// v_wmma_f32_16x16x32_f16) and C stores use immediate offsets. Wt staged in
// LDS; each lane's B fragment is one contiguous 32B LDS read (2x ds_load_b128).
// M multiple of 16 (50000 = 3125*16). Tile guard is wave-uniform -> EXEC all
// ones around WMMA as the ISA requires.
// ---------------------------------------------------------------------------
template <int K, int NOUT>
__global__ void __launch_bounds__(256)
k_gemm_wmma(const _Float16* __restrict__ A, const _Float16* __restrict__ Wt,
            float* __restrict__ C, int M) {
    constexpr int STRIDE = K + WPAD;
    __shared__ _Float16 sW[NOUT * STRIDE];

    for (int i = threadIdx.x; i < NOUT * STRIDE; i += 256) sW[i] = Wt[i];
    __syncthreads();

    int wave = threadIdx.x >> 5;
    int lane = threadIdx.x & 31;
    constexpr int tilesN = NOUT / 16;
    int tile = blockIdx.x * 8 + wave;
    int tM = tile / tilesN;
    int tN = tile - tM * tilesN;
    if (tM >= (M >> 4)) return;              // wave-uniform exit

    // A fragment (16-bit A 16x32, contiguous k order shared with B):
    //   lanes 0-15 : row = lane,    halfs = K[kb+0..15]
    //   lanes 16-31: row = lane-16, halfs = K[kb+16..31]
    // B fragment (16-bit B 32x16):
    //   lanes 0-15 : col = lane,    halfs = K[kb+0..15]   (contiguous in sW)
    //   lanes 16-31: col = lane-16, halfs = K[kb+16..31]
    int row  = (tM << 4) + (lane & 15);
    int col  = (tN << 4) + (lane & 15);
    const _Float16* Arow = A + (size_t)row * K;
    const _Float16* Bcol = sW + col * STRIDE;
    int aoff = (lane < 16) ? 0 : 16;
    int boff = (lane < 16) ? 0 : 16;

    v8f c = {};
#pragma unroll
    for (int kb = 0; kb < K; kb += 32) {
        v8h lo = *(const v8h*)(Arow + kb + aoff);        // 16B-aligned b128 loads
        v8h hi = *(const v8h*)(Arow + kb + aoff + 8);
        v16h a;
#pragma unroll
        for (int i = 0; i < 8; ++i) { a[i] = lo[i]; a[8 + i] = hi[i]; }

        v16h b = *(const v16h*)(Bcol + kb + boff);       // 32B contiguous LDS read

        c = __builtin_amdgcn_wmma_f32_16x16x32_f16(false, a, false, b,
                                                   (short)0, c, false, false);
    }

    // C layout: VGPR r -> M = r (lanes 0-15) / 8+r (lanes 16-31), N = lane&15.
    // Single base address + compile-time immediate offsets.
    int r0 = (tM << 4) + ((lane < 16) ? 0 : 8);
    float* Cb = C + (size_t)r0 * NOUT + col;
#pragma unroll
    for (int r = 0; r < 8; ++r)
        Cb[r * NOUT] = c[r];
}

// ---------------------------------------------------------------------------
// Edge scatter-max: one wave32 per edge (grid-stride), lane-per-feature so the
// gather of H[src] and the atomic-max into out[dst] are coalesced. norm==0
// messages still participate in the max (msg = 0), so no skipping.
// ---------------------------------------------------------------------------
__global__ void __launch_bounds__(256)
k_scatter_max(const float* __restrict__ H, const float* __restrict__ nrm,
              const int* __restrict__ src, const int* __restrict__ dst,
              float* __restrict__ out, int e, int F) {
    int gw   = (int)((blockIdx.x * blockDim.x + threadIdx.x) >> 5);
    int lane = threadIdx.x & 31;
    int nw   = (int)((gridDim.x * blockDim.x) >> 5);
    for (int i = gw; i < e; i += nw) {
        float w = nrm[i];
        const float* hs = H + (size_t)src[i] * F;
        float* od = out + (size_t)dst[i] * F;
        if (i + nw < e)                       // global_prefetch next source row
            __builtin_prefetch(H + (size_t)src[i + nw] * F + lane, 0, 0);
        for (int f = lane; f < F; f += 32)
            atomic_max_f32(&od[f], w * hs[f]);
    }
}

// ---------------------------------------------------------------------------
// Finalize: fold self-loop message into the max, zero non-finite (empty
// segment guard), add bias; optionally emit f16 copy for the next WMMA GEMM.
// ---------------------------------------------------------------------------
__global__ void k_finalize(float* __restrict__ S, const float* __restrict__ H,
                           const float* __restrict__ selfnorm, const float* __restrict__ bias,
                           _Float16* __restrict__ outh, int n, int F) {
    int i = blockIdx.x * blockDim.x + threadIdx.x;
    if (i >= n * F) return;
    int node = i / F;
    int f    = i - node * F;
    float v  = fmaxf(S[i], selfnorm[node] * H[i]);
    unsigned u = __float_as_uint(v);
    if ((u & 0x7F800000u) == 0x7F800000u) v = 0.0f;   // !isfinite -> 0
    v += bias[f];
    S[i] = v;
    if (outh) outh[i] = (_Float16)v;
}

// ---------------------------------------------------------------------------
extern "C" void kernel_launch(void* const* d_in, const int* in_sizes, int n_in,
                              void* d_out, int out_size, void* d_ws, size_t ws_size,
                              hipStream_t stream) {
    const float* x[2]  = { (const float*)d_in[0], (const float*)d_in[3] };
    const int*   ei[2] = { (const int*)d_in[1],   (const int*)d_in[4]   };
    const float* ew[2] = { (const float*)d_in[2], (const float*)d_in[5] };
    const float* W1 = (const float*)d_in[6];
    const float* b1 = (const float*)d_in[7];
    const float* W2 = (const float*)d_in[8];
    const float* b2 = (const float*)d_in[9];
    float* out = (float*)d_out;

    // Workspace carve-up (256B aligned slices)
    char*  ws  = (char*)d_ws;
    size_t off = 0;
    auto carve = [&](size_t bytes) -> void* {
        void* p = ws + off;
        off += (bytes + 255) & ~(size_t)255;
        return p;
    };
    float*    deg      = (float*)   carve((size_t)NN * 4);
    float*    dinv     = (float*)   carve((size_t)NN * 4);
    float*    selfnorm = (float*)   carve((size_t)NN * 4);
    float*    nrm      = (float*)   carve((size_t)EE * 4);
    _Float16* Xh       = (_Float16*)carve((size_t)NN * IN_DIM * 2);
    _Float16* W1t      = (_Float16*)carve((size_t)OUT1 * (IN_DIM + WPAD) * 2);
    _Float16* W2t      = (_Float16*)carve((size_t)OUT2 * (OUT1 + WPAD) * 2);
    float*    H1       = (float*)   carve((size_t)NN * OUT1 * 4);
    float*    S1       = (float*)   carve((size_t)NN * OUT1 * 4);
    _Float16* O1h      = (_Float16*)carve((size_t)NN * OUT1 * 2);
    float*    H2       = (float*)   carve((size_t)NN * OUT2 * 4);

    const int BS = 256;
    auto blk = [](int n) { return (n + 255) / 256; };
    const float NEG_INF = __builtin_huge_valf() * -1.0f;

    // Transpose + convert shared weights once (B-operand layout)
    k_w_transpose_f16<<<blk(OUT1 * (IN_DIM + WPAD)), BS, 0, stream>>>(W1, W1t, IN_DIM, OUT1);
    k_w_transpose_f16<<<blk(OUT2 * (OUT1 + WPAD)), BS, 0, stream>>>(W2, W2t, OUT1, OUT2);

    for (int t = 0; t < 2; ++t) {
        const int* src = ei[t];
        const int* dst = ei[t] + EE;
        float* outT = out + (size_t)t * NN * OUT2;

        // Normalization: deg = 2.0 (self loop) + sum relu(ew) at dst
        k_fill_f32<<<blk(NN), BS, 0, stream>>>(deg, FILLW, NN);
        k_accum_deg<<<blk(EE), BS, 0, stream>>>(dst, ew[t], deg, EE);
        k_dinv_self<<<blk(NN), BS, 0, stream>>>(deg, dinv, selfnorm, NN);
        k_edge_norm<<<blk(EE), BS, 0, stream>>>(src, dst, ew[t], dinv, nrm, EE);

        // Layer 1: H1 = X @ W1 (WMMA), scatter-max, finalize (+f16 copy)
        k_f32_to_f16<<<blk(NN * IN_DIM), BS, 0, stream>>>(x[t], Xh, NN * IN_DIM);
        {
            int tiles = (NN / 16) * (OUT1 / 16);
            k_gemm_wmma<IN_DIM, OUT1><<<(tiles + 7) / 8, BS, 0, stream>>>(Xh, W1t, H1, NN);
        }
        k_fill_f32<<<blk(NN * OUT1), BS, 0, stream>>>(S1, NEG_INF, NN * OUT1);
        k_scatter_max<<<8192, BS, 0, stream>>>(H1, nrm, src, dst, S1, EE, OUT1);
        k_finalize<<<blk(NN * OUT1), BS, 0, stream>>>(S1, H1, selfnorm, b1, O1h, NN, OUT1);

        // Layer 2: H2 = O1 @ W2 (WMMA), scatter-max into d_out slice, finalize
        {
            int tiles = (NN / 16) * (OUT2 / 16);
            k_gemm_wmma<OUT1, OUT2><<<(tiles + 7) / 8, BS, 0, stream>>>(O1h, W2t, H2, NN);
        }
        k_fill_f32<<<blk(NN * OUT2), BS, 0, stream>>>(outT, NEG_INF, NN * OUT2);
        k_scatter_max<<<8192, BS, 0, stream>>>(H2, nrm, src, dst, outT, EE, OUT2);
        k_finalize<<<blk(NN * OUT2), BS, 0, stream>>>(outT, H2, selfnorm, b2, (_Float16*)nullptr, NN, OUT2);
    }
}
